// ConditionalDDPM_15728170238302
// MI455X (gfx1250) — matrix operand order repaired and checked by reading, not compiled, not run on hardware
//
#include <hip/hip_runtime.h>

// ---------------------------------------------------------------------------
// CDNA5 (gfx1250) conditional-DDPM UNet forward + MSE.
// Conv / convT / dense math via v_wmma_f32_16x16x32_f16 (f16 in, f32 accum),
// implicit-im2col staged through LDS, pair-interleaved packed weights,
// async global->LDS weight staging (ASYNCcnt), 128x64 block tile
// (8 wave32, 32x32 per wave => 4 WMMA / K-step).
// ---------------------------------------------------------------------------

typedef __attribute__((ext_vector_type(16))) _Float16 v16h;
typedef __attribute__((ext_vector_type(8)))  float    v8f;
typedef _Float16 f16;

#define BATCH   512
#define NF      128
#define HW28    784
#define HW14    196
#define HW7     49

__device__ __forceinline__ float gelu_f(float v) {
    float v3 = v * v * v;
    return 0.5f * v * (1.0f + tanhf(0.7978845608028654f * (v + 0.044715f * v3)));
}

union FragH { v16h h; unsigned int u[8]; };

// ---------------------------------------------------------------------------
// Templated WMMA GEMM kernel (only 3 instantiations are used):
//   AMODE 0 = A flat row-major f16 [M][K]            (OMODE 0: f32 [M][N])
//   AMODE 1 = implicit im2col 3x3 SAME, concat a0/a1 (OMODE 1: f32 NCHW)
//   AMODE 2 = channel gather at pixel                (OMODE 2: f16 convT scatter)
// Weights pair-interleaved: wp[k/2][Npack][2] (zero padded, Npack = ceil64(N)).
// ---------------------------------------------------------------------------
template<int AMODE, int OMODE>
__global__ __launch_bounds__(256)
void wmma_gemm_t(const f16* __restrict__ Aflat,
                 const f16* __restrict__ a0, int C0,
                 const f16* __restrict__ a1, int C1,
                 int H, int W,
                 const f16* __restrict__ wp, int Npack,
                 const float* __restrict__ bias, int bias_div,
                 int M, int K, int Kpad, int N,
                 float* __restrict__ outF, f16* __restrict__ outH)
{
    __shared__ __align__(16) unsigned short As[128 * 32];      // [m][kk]
    __shared__ __align__(16) unsigned short Bs[16 * 64 * 2];   // [kp][n][2]

    const int tid  = threadIdx.x;
    const int lane = tid & 31;
    const int wave = tid >> 5;
    const int rt   = wave & 3;          // 32-row group
    const int ct   = wave >> 2;         // 32-col group
    const int mBase = blockIdx.x * 128;
    const int nBase = blockIdx.y * 64;
    const int par   = blockIdx.z;
    const int py = par >> 1, px = par & 1;
    const int HW = H * W;
    const f16* wpp = wp + (OMODE == 2 ? (size_t)par * (Kpad / 2) * Npack * 2 : 0);

    // per-thread fixed k-lane; incremental (cin, rem) decomposition for im2col
    const int kk = tid & 31;
    int cin = kk / 9;                  // one-time division
    int rem = kk - cin * 9;

    // per-slot row geometry, hoisted out of the K loop
    int base[16];      // AMODE0: row*K ; AMODE1/2: b*HW
    int g[16];         // AMODE1: (y<<10)|x ; AMODE2: pixel p
    unsigned rowmask = 0;
    #pragma unroll
    for (int i = 0; i < 16; ++i) {
        int m   = (tid >> 5) + i * 8;
        int row = mBase + m;
        int ok  = (row < M) ? 1 : 0;
        int rr  = ok ? row : 0;
        if (AMODE == 0) {
            base[i] = rr * K;
            g[i] = 0;
        } else {
            int b = rr / HW, p = rr - b * HW;
            base[i] = b * HW;
            if (AMODE == 2) {
                g[i] = p;
            } else {
                int y = p / W, x = p - y * W;
                g[i] = (y << 10) | x;
            }
        }
        rowmask |= (ok ? (1u << i) : 0u);
    }

    v8f acc00 = {}, acc01 = {}, acc10 = {}, acc11 = {};

    const int bkp = tid >> 4;          // 0..15 : k-pair row in B tile
    const int bq  = tid & 15;          // 0..15 : 16B chunk within 256B row
    // LDS byte offset of this thread's 16B destination chunk in Bs
    const unsigned bsOff = (unsigned)(uintptr_t)(&Bs[0]) + (unsigned)tid * 16u;

    for (int kb = 0; kb < Kpad; kb += 32) {
        const int k = kb + kk;
        const bool kvalid = k < K;

        // ---- per-step k decomposition (no divisions) ----
        const f16* src = a0;
        int Cc = C0, ccHW = 0, dy = 0, dx = 0;
        if (AMODE == 1) {
            dy = (rem >= 6) ? 1 : ((rem >= 3) ? 0 : -1);
            dx = rem - (dy + 1) * 3 - 1;
            int cc = cin;
            if (cin >= C0) { src = a1; cc = cin - C0; Cc = C1; }
            ccHW = cc * HW;
        } else if (AMODE == 2) {
            int cc = k;
            if (k >= C0) { src = a1; cc = k - C0; Cc = C1; }
            ccHW = cc * HW;
        }

        // ---- async-stage B tile (32k x 64n pair-interleaved, 16B/lane) ----
        {
            const f16* srow = wpp + ((size_t)((kb >> 1) + bkp) * Npack + nBase) * 2;
            const void* gaddr = (const void*)(reinterpret_cast<const uint4*>(srow) + bq);
            asm volatile("global_load_async_to_lds_b128 %0, %1, off"
                         :: "v"(bsOff), "v"(gaddr) : "memory");
            if (kb + 32 < Kpad) {
                const f16* nrow = wpp + ((size_t)((kb >> 1) + 16 + bkp) * Npack + nBase) * 2;
                __builtin_prefetch(reinterpret_cast<const uint4*>(nrow) + bq, 0, 3);
            }
        }

        // ---- stage A tile (128 x 32): unconditional clamped loads + select ----
        #pragma unroll
        for (int i = 0; i < 16; ++i) {
            bool rv = kvalid && ((rowmask >> i) & 1u);
            f16 hv;
            if (AMODE == 0) {
                int idx = base[i] + k;
                hv = Aflat[rv ? idx : 0];
                hv = rv ? hv : (f16)0.0f;
            } else if (AMODE == 1) {
                int y = (g[i] >> 10) + dy;
                int x = (g[i] & 1023) + dx;
                bool ok = rv && ((unsigned)y < (unsigned)H) && ((unsigned)x < (unsigned)W);
                int idx = base[i] * Cc + ccHW + y * W + x;
                hv = src[ok ? idx : 0];
                hv = ok ? hv : (f16)0.0f;
            } else {
                int idx = base[i] * Cc + ccHW + g[i];
                hv = src[rv ? idx : 0];
                hv = rv ? hv : (f16)0.0f;
            }
            As[(tid >> 5) * 32 + kk + i * 256] = *reinterpret_cast<unsigned short*>(&hv);
        }

        // async B copy must be complete before the barrier releases readers
        asm volatile("s_wait_asynccnt 0x0" ::: "memory");
        __syncthreads();

        // ---- fragments (ISA 16-bit A layout / pair-interleaved B) ----
        FragH fa0, fa1, fb0, fb1;
        {
            int m0 = rt * 32 + (lane & 15);
            int kh = (lane >> 4) * 8;
            #pragma unroll
            for (int j = 0; j < 8; ++j) {
                int kf = (j < 4) ? (kh + 2 * j) : (kh + 16 + 2 * (j - 4));
                fa0.u[j] = *reinterpret_cast<const unsigned int*>(&As[m0 * 32 + kf]);
                fa1.u[j] = *reinterpret_cast<const unsigned int*>(&As[(m0 + 16) * 32 + kf]);
            }
            int n0 = ct * 32 + (lane & 15);
            #pragma unroll
            for (int j = 0; j < 8; ++j) {
                int kp = (j < 4) ? ((lane >> 4) * 4 + j) : (8 + (lane >> 4) * 4 + (j - 4));
                fb0.u[j] = *reinterpret_cast<const unsigned int*>(&Bs[(kp * 64 + n0) * 2]);
                fb1.u[j] = *reinterpret_cast<const unsigned int*>(&Bs[(kp * 64 + n0 + 16) * 2]);
            }
        }
        acc00 = __builtin_amdgcn_wmma_f32_16x16x32_f16(false, fa0.h, false, fb0.h, (short)0, acc00, false, false);
        acc01 = __builtin_amdgcn_wmma_f32_16x16x32_f16(false, fa0.h, false, fb1.h, (short)0, acc01, false, false);
        acc10 = __builtin_amdgcn_wmma_f32_16x16x32_f16(false, fa1.h, false, fb0.h, (short)0, acc10, false, false);
        acc11 = __builtin_amdgcn_wmma_f32_16x16x32_f16(false, fa1.h, false, fb1.h, (short)0, acc11, false, false);

        if (AMODE == 1) {              // advance incremental decomposition
            cin += 3; rem += 5;
            if (rem >= 9) { rem -= 9; ++cin; }
        }
        __syncthreads();
    }

    // ---- epilogue (C layout: VGPR j -> row j + 8*(lane>=16); col = lane&15) ----
    #pragma unroll
    for (int mt = 0; mt < 2; ++mt) {
        #pragma unroll
        for (int t = 0; t < 2; ++t) {
            const v8f acc = mt ? (t ? acc11 : acc10) : (t ? acc01 : acc00);
            int col = nBase + ct * 32 + t * 16 + (lane & 15);
            if (col >= N) continue;
            float bv = bias ? bias[col / bias_div] : 0.0f;
            #pragma unroll
            for (int j = 0; j < 8; ++j) {
                int row = mBase + rt * 32 + mt * 16 + (lane >> 4) * 8 + j;
                if (row >= M) continue;
                float v = acc[j] + bv;
                if (OMODE == 0) {
                    outF[(size_t)row * N + col] = v;
                } else if (OMODE == 1) {
                    int b = row / HW, p = row - b * HW;
                    outF[((size_t)b * N + col) * HW + p] = v;
                } else {
                    int b = row / HW, p = row - b * HW;
                    int y = p / W,  x = p - y * W;
                    int oy = 2 * y + py, ox = 2 * x + px;
                    outH[(((size_t)b * N + col) * (2 * H) + oy) * (2 * W) + ox] = (f16)v;
                }
            }
        }
    }
}

// ---------------------------------------------------------------------------
// GroupNorm (8 groups) + activation, f32 raw -> f16. One block per (b, g).
// act: 0 = none, 1 = gelu, 2 = relu
// ---------------------------------------------------------------------------
__global__ __launch_bounds__(256)
void gn_act_kernel(const float* __restrict__ x, int C, int HW,
                   const float* __restrict__ gamma, const float* __restrict__ beta,
                   int act, f16* __restrict__ out)
{
    __shared__ float s1[256], s2[256];
    int b = blockIdx.x >> 3, g = blockIdx.x & 7;
    int cpg = C >> 3;
    int n = cpg * HW;
    const float* base = x + ((size_t)b * C + g * cpg) * HW;
    float a = 0.0f, q = 0.0f;
    for (int i = threadIdx.x; i < n; i += 256) {
        float v = base[i]; a += v; q += v * v;
    }
    s1[threadIdx.x] = a; s2[threadIdx.x] = q;
    __syncthreads();
    for (int s = 128; s > 0; s >>= 1) {
        if (threadIdx.x < s) {
            s1[threadIdx.x] += s1[threadIdx.x + s];
            s2[threadIdx.x] += s2[threadIdx.x + s];
        }
        __syncthreads();
    }
    float mu  = s1[0] / (float)n;
    float var = s2[0] / (float)n - mu * mu;
    float rs  = rsqrtf(var + 1e-5f);
    f16* ob = out + ((size_t)b * C + g * cpg) * HW;
    for (int i = threadIdx.x; i < n; i += 256) {
        int c = g * cpg + i / HW;
        float v = (base[i] - mu) * rs * gamma[c] + beta[c];
        if (act == 1)      v = gelu_f(v);
        else if (act == 2) v = fmaxf(v, 0.0f);
        ob[i] = (f16)v;
    }
}

// --------------------------- small prep / elementwise -----------------------

__global__ void prep_kernel(const int* __restrict__ t_idx, const int* __restrict__ cond,
                            const float* __restrict__ keep,
                            float* sa, float* s1m, float* tn, float* cvec)
{
    int b = blockIdx.x * blockDim.x + threadIdx.x;
    if (b >= BATCH) return;
    int t = t_idx[b];
    float ab = 1.0f;
    for (int i = 0; i <= t; ++i) {
        float beta = 1e-4f + (0.02f - 1e-4f) * (float)i / 399.0f;
        ab *= (1.0f - beta);
    }
    sa[b]  = sqrtf(ab);
    s1m[b] = sqrtf(1.0f - ab);
    tn[b]  = ((float)t + 1.0f) / 400.0f;
    int c = cond[b];
    float km = keep[b];
    for (int j = 0; j < 10; ++j)
        cvec[b * 10 + j] = (km == 1.0f) ? ((j == c) ? 1.0f : 0.0f) : -1.0f;
}

__global__ void noise_mix_kernel(const float* __restrict__ img, const float* __restrict__ nz,
                                 const float* __restrict__ sa, const float* __restrict__ s1m,
                                 f16* __restrict__ out)
{
    int i = blockIdx.x * 256 + threadIdx.x;
    if (i >= BATCH * HW28) return;
    int b = i / HW28;
    out[i] = (f16)(sa[b] * img[i] + s1m[b] * nz[i]);
}

__global__ void dense_kernel(const float* __restrict__ in, int din,
                             const float* __restrict__ w, const float* __restrict__ bias,
                             int dout, int act, float* __restrict__ out)
{
    int i = blockIdx.x * 256 + threadIdx.x;
    if (i >= BATCH * dout) return;
    int b = i / dout, j = i - b * dout;
    float s = bias[j];
    for (int k = 0; k < din; ++k) s += in[b * din + k] * w[k * dout + j];
    if (act) s = gelu_f(s);
    out[i] = s;
}

__global__ void maxpool2_kernel(const f16* __restrict__ in, f16* __restrict__ out,
                                int C, int H, int W)
{
    int Ho = H >> 1, Wo = W >> 1;
    long long total = (long long)BATCH * C * Ho * Wo;
    long long i = (long long)blockIdx.x * 256 + threadIdx.x;
    if (i >= total) return;
    int x = (int)(i % Wo); long long r = i / Wo;
    int y = (int)(r % Ho); r /= Ho;
    int c = (int)(r % C);  int b = (int)(r / C);
    const f16* base = in + (((size_t)b * C + c) * H + 2 * y) * W + 2 * x;
    float v0 = (float)base[0], v1 = (float)base[1];
    float v2 = (float)base[W], v3 = (float)base[W + 1];
    out[i] = (f16)fmaxf(fmaxf(v0, v1), fmaxf(v2, v3));
}

__global__ void meanpool_gelu_kernel(const f16* __restrict__ d2, f16* __restrict__ hv)
{
    int i = blockIdx.x * 256 + threadIdx.x;           // (b, c) over 512*256
    if (i >= BATCH * 256) return;
    const f16* base = d2 + (size_t)i * HW7;
    float s = 0.0f;
    for (int k = 0; k < HW7; ++k) s += (float)base[k];
    hv[i] = (f16)gelu_f(s / (float)HW7);
}

__global__ void mix_kernel(const f16* __restrict__ u, const float* __restrict__ ce,
                           const float* __restrict__ te, int C, int HW,
                           f16* __restrict__ out)
{
    long long total = (long long)BATCH * C * HW;
    long long i = (long long)blockIdx.x * 256 + threadIdx.x;
    if (i >= total) return;
    int bc = (int)(i / HW);
    out[i] = (f16)(ce[bc] * (float)u[i] + te[bc]);
}

// --------------------------- weight packing ---------------------------------
// All GEMM B matrices packed pair-interleaved: wp[k/2][Npack][2], zero padded.

__global__ void pack_conv3_kernel(const float* __restrict__ w, int Cin, int Cout,
                                  int Kpad, int Npack, f16* __restrict__ wp)
{
    int i = blockIdx.x * 256 + threadIdx.x;
    if (i >= Kpad * Npack) return;
    int n2 = Npack * 2;
    int kp = i / n2, r2 = i - kp * n2;
    int n = r2 >> 1, r = r2 & 1;
    int k = kp * 2 + r;
    float v = 0.0f;
    if (k < Cin * 9 && n < Cout) {
        int cin = k / 9, rem = k - cin * 9;
        v = w[((size_t)n * Cin + cin) * 9 + rem];
    }
    wp[i] = (f16)v;
}

__global__ void pack_convT_kernel(const float* __restrict__ w, int Cin, int Cout,
                                  int Npack, f16* __restrict__ wp)
{
    int i = blockIdx.x * 256 + threadIdx.x;
    if (i >= 4 * Cin * Npack) return;
    int per = Cin * Npack;
    int p = i / per, rr = i - p * per;
    int n2 = Npack * 2;
    int kp = rr / n2, r2 = rr - kp * n2;
    int n = r2 >> 1, r = r2 & 1;
    int ci = kp * 2 + r;
    int py = p >> 1, px = p & 1;
    float v = 0.0f;
    if (n < Cout) v = w[(((size_t)ci * Cout + n) * 2 + py) * 2 + px];
    wp[i] = (f16)v;
}

__global__ void pack_gemmB_kernel(const float* __restrict__ w, int K, int N,
                                  int Kpad, int Npack, f16* __restrict__ wp)
{
    long long i = (long long)blockIdx.x * 256 + threadIdx.x;
    if (i >= (long long)Kpad * Npack) return;
    int n2 = Npack * 2;
    int kp = (int)(i / n2), r2 = (int)(i - (long long)kp * n2);
    int n = r2 >> 1, r = r2 & 1;
    int k = kp * 2 + r;
    float v = 0.0f;
    if (k < K && n < N) v = w[(size_t)k * N + n];
    wp[i] = (f16)v;
}

// --------------------------- loss reduction ---------------------------------

__global__ __launch_bounds__(256)
void loss1_kernel(const float* __restrict__ pred, const float* __restrict__ nz,
                  float* __restrict__ part)
{
    __shared__ float s[256];
    const long long N = (long long)BATCH * HW28;
    float a = 0.0f;
    for (long long i = (long long)blockIdx.x * 256 + threadIdx.x; i < N; i += 256LL * 256) {
        float d = pred[i] - nz[i];
        a += d * d;
    }
    s[threadIdx.x] = a;
    __syncthreads();
    for (int k = 128; k > 0; k >>= 1) {
        if (threadIdx.x < k) s[threadIdx.x] += s[threadIdx.x + k];
        __syncthreads();
    }
    if (threadIdx.x == 0) part[blockIdx.x] = s[0];
}

__global__ __launch_bounds__(256)
void loss2_kernel(const float* __restrict__ part, float* __restrict__ out)
{
    __shared__ float s[256];
    s[threadIdx.x] = part[threadIdx.x];
    __syncthreads();
    for (int k = 128; k > 0; k >>= 1) {
        if (threadIdx.x < k) s[threadIdx.x] += s[threadIdx.x + k];
        __syncthreads();
    }
    if (threadIdx.x == 0) out[0] = s[0] / (float)((long long)BATCH * HW28);
}

// ---------------------------------------------------------------------------
// Host launcher
// ---------------------------------------------------------------------------
static inline unsigned cdiv(long long a, long long b) { return (unsigned)((a + b - 1) / b); }

extern "C" void kernel_launch(void* const* d_in, const int* in_sizes, int n_in,
                              void* d_out, int out_size, void* d_ws, size_t ws_size,
                              hipStream_t stream)
{
    (void)in_sizes; (void)n_in; (void)out_size; (void)ws_size;

    // Params dict flattened in sorted-key (jax tree) order, then the 5 arrays.
    const float* cemb1_b1 = (const float*)d_in[0];
    const float* cemb1_b2 = (const float*)d_in[1];
    const float* cemb1_w1 = (const float*)d_in[2];
    const float* cemb1_w2 = (const float*)d_in[3];
    const float* cemb2_b1 = (const float*)d_in[4];
    const float* cemb2_b2 = (const float*)d_in[5];
    const float* cemb2_w1 = (const float*)d_in[6];
    const float* cemb2_w2 = (const float*)d_in[7];
    const float* down1_b  = (const float*)d_in[8];
    const float* down1_be = (const float*)d_in[9];
    const float* down1_g  = (const float*)d_in[10];
    const float* down1_w  = (const float*)d_in[11];
    const float* down2_b  = (const float*)d_in[12];
    const float* down2_be = (const float*)d_in[13];
    const float* down2_g  = (const float*)d_in[14];
    const float* down2_w  = (const float*)d_in[15];
    const float* init_b   = (const float*)d_in[16];
    const float* init_be  = (const float*)d_in[17];
    const float* init_g   = (const float*)d_in[18];
    const float* init_w   = (const float*)d_in[19];
    const float* out1_b   = (const float*)d_in[20];
    const float* out1_be  = (const float*)d_in[21];
    const float* out1_g   = (const float*)d_in[22];
    const float* out1_w   = (const float*)d_in[23];
    const float* out2_b   = (const float*)d_in[24];
    const float* out2_w   = (const float*)d_in[25];
    const float* temb1_b1 = (const float*)d_in[26];
    const float* temb1_b2 = (const float*)d_in[27];
    const float* temb1_w1 = (const float*)d_in[28];
    const float* temb1_w2 = (const float*)d_in[29];
    const float* temb2_b1 = (const float*)d_in[30];
    const float* temb2_b2 = (const float*)d_in[31];
    const float* temb2_w1 = (const float*)d_in[32];
    const float* temb2_w2 = (const float*)d_in[33];
    const float* up0_b    = (const float*)d_in[34];
    const float* up0_be   = (const float*)d_in[35];
    const float* up0_g    = (const float*)d_in[36];
    const float* up0_w    = (const float*)d_in[37];
    const float* up1c_b   = (const float*)d_in[38];
    const float* up1c_be  = (const float*)d_in[39];
    const float* up1c_g   = (const float*)d_in[40];
    const float* up1c_w   = (const float*)d_in[41];
    const float* up1t_b   = (const float*)d_in[42];
    const float* up1t_w   = (const float*)d_in[43];
    const float* up2c_b   = (const float*)d_in[44];
    const float* up2c_be  = (const float*)d_in[45];
    const float* up2c_g   = (const float*)d_in[46];
    const float* up2c_w   = (const float*)d_in[47];
    const float* up2t_b   = (const float*)d_in[48];
    const float* up2t_w   = (const float*)d_in[49];
    const float* images   = (const float*)d_in[50];
    const float* noise    = (const float*)d_in[51];
    const float* keep     = (const float*)d_in[52];
    const int*   t_idx    = (const int*)d_in[53];
    const int*   conds    = (const int*)d_in[54];

    // ---------------- workspace layout (static offsets) --------------------
    char* ws = (char*)d_ws;
    size_t off = 0;
    auto alloc = [&](size_t bytes) { size_t r = off; off += (bytes + 255) & ~(size_t)255; return r; };

    float* SA   = (float*)(ws + alloc(BATCH * 4));
    float* S1M  = (float*)(ws + alloc(BATCH * 4));
    float* TN   = (float*)(ws + alloc(BATCH * 4));
    float* CVEC = (float*)(ws + alloc(BATCH * 10 * 4));
    float* HTMP = (float*)(ws + alloc((size_t)BATCH * 256 * 4));
    float* TE1  = (float*)(ws + alloc((size_t)BATCH * 256 * 4));
    float* TE2  = (float*)(ws + alloc((size_t)BATCH * 128 * 4));
    float* CE1  = (float*)(ws + alloc((size_t)BATCH * 256 * 4));
    float* CE2  = (float*)(ws + alloc((size_t)BATCH * 128 * 4));
    f16*  X16   = (f16*)(ws + alloc((size_t)BATCH * HW28 * 2));
    f16*  WP    = (f16*)(ws + alloc((size_t)8 * 1024 * 1024));           // packed weights
    float* RAW  = (float*)(ws + alloc((size_t)BATCH * NF * HW28 * 4));   // reused
    f16*  H0    = (f16*)(ws + alloc((size_t)BATCH * NF * HW28 * 2));     // skip
    f16*  BUFA  = (f16*)(ws + alloc((size_t)BATCH * NF * HW28 * 2));     // a1 -> u2pre -> h
    f16*  BUFB  = (f16*)(ws + alloc((size_t)BATCH * NF * HW28 * 2));     // a2 -> u2
    f16*  D1    = (f16*)(ws + alloc((size_t)BATCH * NF * HW14 * 2));
    f16*  D2    = (f16*)(ws + alloc((size_t)BATCH * 256 * HW7 * 2));
    f16*  HV    = (f16*)(ws + alloc((size_t)BATCH * 256 * 2));
    f16*  U0    = (f16*)(ws + alloc((size_t)BATCH * 256 * HW7 * 2));
    f16*  MIX1  = (f16*)(ws + alloc((size_t)BATCH * 256 * HW7 * 2));
    f16*  U1P   = (f16*)(ws + alloc((size_t)BATCH * NF * HW14 * 2));
    f16*  U1    = (f16*)(ws + alloc((size_t)BATCH * NF * HW14 * 2));
    f16*  MIX2  = (f16*)(ws + alloc((size_t)BATCH * NF * HW14 * 2));
    float* PART = (float*)(ws + alloc(256 * 4));

    const dim3 blk(256);
    const long long M28 = (long long)BATCH * HW28;   // 401408
    const long long M14 = (long long)BATCH * HW14;   // 100352
    const long long M7  = (long long)BATCH * HW7;    // 25088

    // ---- scalars + noised input ----
    prep_kernel<<<2, blk, 0, stream>>>(t_idx, conds, keep, SA, S1M, TN, CVEC);
    noise_mix_kernel<<<cdiv(M28, 256), blk, 0, stream>>>(images, noise, SA, S1M, X16);

    // ---- tiny MLP embeddings ----
    dense_kernel<<<cdiv((long long)BATCH * 256, 256), blk, 0, stream>>>(TN, 1, temb1_w1, temb1_b1, 256, 1, HTMP);
    dense_kernel<<<cdiv((long long)BATCH * 256, 256), blk, 0, stream>>>(HTMP, 256, temb1_w2, temb1_b2, 256, 0, TE1);
    dense_kernel<<<cdiv((long long)BATCH * 128, 256), blk, 0, stream>>>(TN, 1, temb2_w1, temb2_b1, 128, 1, HTMP);
    dense_kernel<<<cdiv((long long)BATCH * 128, 256), blk, 0, stream>>>(HTMP, 128, temb2_w2, temb2_b2, 128, 0, TE2);
    dense_kernel<<<cdiv((long long)BATCH * 256, 256), blk, 0, stream>>>(CVEC, 10, cemb1_w1, cemb1_b1, 256, 1, HTMP);
    dense_kernel<<<cdiv((long long)BATCH * 256, 256), blk, 0, stream>>>(HTMP, 256, cemb1_w2, cemb1_b2, 256, 0, CE1);
    dense_kernel<<<cdiv((long long)BATCH * 128, 256), blk, 0, stream>>>(CVEC, 10, cemb2_w1, cemb2_b1, 128, 1, HTMP);
    dense_kernel<<<cdiv((long long)BATCH * 128, 256), blk, 0, stream>>>(HTMP, 128, cemb2_w2, cemb2_b2, 128, 0, CE2);

    // ---- init: conv 1->128 (K=9 pad 32), GN, GELU -> H0 ----
    pack_conv3_kernel<<<cdiv(32 * 128, 256), blk, 0, stream>>>(init_w, 1, 128, 32, 128, WP);
    wmma_gemm_t<1, 1><<<dim3(cdiv(M28, 128), 2, 1), blk, 0, stream>>>(
        nullptr, X16, 1, nullptr, 0, 28, 28, WP, 128, init_b, 1,
        (int)M28, 9, 32, 128, RAW, nullptr);
    gn_act_kernel<<<BATCH * 8, blk, 0, stream>>>(RAW, 128, HW28, init_g, init_be, 1, H0);

    // ---- down1 ----
    pack_conv3_kernel<<<cdiv(1152 * 128, 256), blk, 0, stream>>>(down1_w, 128, 128, 1152, 128, WP);
    wmma_gemm_t<1, 1><<<dim3(cdiv(M28, 128), 2, 1), blk, 0, stream>>>(
        nullptr, H0, 128, nullptr, 0, 28, 28, WP, 128, down1_b, 1,
        (int)M28, 1152, 1152, 128, RAW, nullptr);
    gn_act_kernel<<<BATCH * 8, blk, 0, stream>>>(RAW, 128, HW28, down1_g, down1_be, 1, BUFA);
    maxpool2_kernel<<<cdiv((long long)BATCH * 128 * HW14, 256), blk, 0, stream>>>(BUFA, D1, 128, 28, 28);

    // ---- down2 ----
    pack_conv3_kernel<<<cdiv(1152 * 256, 256), blk, 0, stream>>>(down2_w, 128, 256, 1152, 256, WP);
    wmma_gemm_t<1, 1><<<dim3(cdiv(M14, 128), 4, 1), blk, 0, stream>>>(
        nullptr, D1, 128, nullptr, 0, 14, 14, WP, 256, down2_b, 1,
        (int)M14, 1152, 1152, 256, RAW, nullptr);
    gn_act_kernel<<<BATCH * 8, blk, 0, stream>>>(RAW, 256, HW14, down2_g, down2_be, 1, BUFB);
    maxpool2_kernel<<<cdiv((long long)BATCH * 256 * HW7, 256), blk, 0, stream>>>(BUFB, D2, 256, 14, 14);

    // ---- hv ----
    meanpool_gelu_kernel<<<cdiv((long long)BATCH * 256, 256), blk, 0, stream>>>(D2, HV);

    // ---- up0: GEMM M=512,K=256,N=256*49; GN+ReLU ----
    pack_gemmB_kernel<<<cdiv((long long)256 * 12544, 256), blk, 0, stream>>>(up0_w, 256, 12544, 256, 12544, WP);
    wmma_gemm_t<0, 0><<<dim3(4, cdiv(12544, 64), 1), blk, 0, stream>>>(
        HV, nullptr, 0, nullptr, 0, 1, 1, WP, 12544, up0_b, 49,
        BATCH, 256, 256, 12544, RAW, nullptr);
    gn_act_kernel<<<BATCH * 8, blk, 0, stream>>>(RAW, 256, HW7, up0_g, up0_be, 2, U0);

    // ---- mix1 ----
    mix_kernel<<<cdiv((long long)BATCH * 256 * HW7, 256), blk, 0, stream>>>(U0, CE1, TE1, 256, HW7, MIX1);

    // ---- up1t: convT 2x2 s2, concat [mix1, d2] (K=512) ----
    pack_convT_kernel<<<cdiv((long long)4 * 512 * 128, 256), blk, 0, stream>>>(up1t_w, 512, 128, 128, WP);
    wmma_gemm_t<2, 2><<<dim3(cdiv(M7, 128), 2, 4), blk, 0, stream>>>(
        nullptr, MIX1, 256, D2, 256, 7, 7, WP, 128, up1t_b, 1,
        (int)M7, 512, 512, 128, nullptr, U1P);

    // ---- up1c ----
    pack_conv3_kernel<<<cdiv(1152 * 128, 256), blk, 0, stream>>>(up1c_w, 128, 128, 1152, 128, WP);
    wmma_gemm_t<1, 1><<<dim3(cdiv(M14, 128), 2, 1), blk, 0, stream>>>(
        nullptr, U1P, 128, nullptr, 0, 14, 14, WP, 128, up1c_b, 1,
        (int)M14, 1152, 1152, 128, RAW, nullptr);
    gn_act_kernel<<<BATCH * 8, blk, 0, stream>>>(RAW, 128, HW14, up1c_g, up1c_be, 1, U1);

    // ---- mix2 ----
    mix_kernel<<<cdiv((long long)BATCH * 128 * HW14, 256), blk, 0, stream>>>(U1, CE2, TE2, 128, HW14, MIX2);

    // ---- up2t: convT 2x2 s2, concat [mix2, d1] (K=256) ----
    pack_convT_kernel<<<cdiv((long long)4 * 256 * 128, 256), blk, 0, stream>>>(up2t_w, 256, 128, 128, WP);
    wmma_gemm_t<2, 2><<<dim3(cdiv(M14, 128), 2, 4), blk, 0, stream>>>(
        nullptr, MIX2, 128, D1, 128, 14, 14, WP, 128, up2t_b, 1,
        (int)M14, 256, 256, 128, nullptr, BUFA);

    // ---- up2c ----
    pack_conv3_kernel<<<cdiv(1152 * 128, 256), blk, 0, stream>>>(up2c_w, 128, 128, 1152, 128, WP);
    wmma_gemm_t<1, 1><<<dim3(cdiv(M28, 128), 2, 1), blk, 0, stream>>>(
        nullptr, BUFA, 128, nullptr, 0, 28, 28, WP, 128, up2c_b, 1,
        (int)M28, 1152, 1152, 128, RAW, nullptr);
    gn_act_kernel<<<BATCH * 8, blk, 0, stream>>>(RAW, 128, HW28, up2c_g, up2c_be, 1, BUFB);

    // ---- out1: conv concat [u2, h0] ----
    pack_conv3_kernel<<<cdiv(2304 * 128, 256), blk, 0, stream>>>(out1_w, 256, 128, 2304, 128, WP);
    wmma_gemm_t<1, 1><<<dim3(cdiv(M28, 128), 2, 1), blk, 0, stream>>>(
        nullptr, BUFB, 128, H0, 128, 28, 28, WP, 128, out1_b, 1,
        (int)M28, 2304, 2304, 128, RAW, nullptr);
    gn_act_kernel<<<BATCH * 8, blk, 0, stream>>>(RAW, 128, HW28, out1_g, out1_be, 2, BUFA);

    // ---- out2: conv 128->1 ----
    pack_conv3_kernel<<<cdiv(1152 * 64, 256), blk, 0, stream>>>(out2_w, 128, 1, 1152, 64, WP);
    wmma_gemm_t<1, 1><<<dim3(cdiv(M28, 128), 1, 1), blk, 0, stream>>>(
        nullptr, BUFA, 128, nullptr, 0, 28, 28, WP, 64, out2_b, 1,
        (int)M28, 1152, 1152, 1, RAW, nullptr);

    // ---- loss ----
    loss1_kernel<<<256, blk, 0, stream>>>(RAW, noise, PART);
    loss2_kernel<<<1, blk, 0, stream>>>(PART, (float*)d_out);
}
